// SelfAttentionLayer_14680198218156
// MI455X (gfx1250) — matrix operand order repaired and checked by reading, not compiled, not run on hardware
//
#include <hip/hip_runtime.h>

// ---------------------------------------------------------------------------
// Problem dims (fixed by reference): B=4, L=2048, C=H=1024
// ---------------------------------------------------------------------------
#define BB   4
#define LL   2048
#define CC   1024
#define HH   1024
#define MM   (BB * LL)          // 8192 total rows

typedef __attribute__((ext_vector_type(16))) __bf16 v16bf;
typedef __attribute__((ext_vector_type(8)))  __bf16 v8bf;
typedef __attribute__((ext_vector_type(8)))  float  v8f;

__device__ __forceinline__ __bf16 f2bf(float f) {
    unsigned u = __builtin_bit_cast(unsigned, f);
    unsigned r = (u + 0x7FFFu + ((u >> 16) & 1u)) >> 16;   // round-nearest-even
    unsigned short s = (unsigned short)r;
    return __builtin_bit_cast(__bf16, s);
}

// Load a 16-element bf16 fragment as two 16-byte halves; `off2` is the element
// offset of the second half (16 for both the A and the B fragments).
__device__ __forceinline__ v16bf load_frag(const __bf16* __restrict__ p, int off2) {
    v8bf lo = *(const v8bf*)(p);
    v8bf hi = *(const v8bf*)(p + off2);
    v16bf r;
    #pragma unroll
    for (int i = 0; i < 8; ++i) { r[i] = lo[i]; r[8 + i] = hi[i]; }
    return r;
}

__device__ __forceinline__ v16bf load_frag_lds(const __bf16* p, int off2) {
    v8bf lo = *(const v8bf*)(p);
    v8bf hi = *(const v8bf*)(p + off2);
    v16bf r;
    #pragma unroll
    for (int i = 0; i < 8; ++i) { r[i] = lo[i]; r[8 + i] = hi[i]; }
    return r;
}

// Workspace layout (bytes)
#define XB_OFF  (0)                                   // bf16 [8192][1024]  16MB
#define WT_OFF  ((size_t)16 << 20)                    // bf16 [3][1024][1024] 6MB
#define KB_OFF  ((size_t)22 << 20)                    // bf16 [8192][1024]  16MB
#define QB_OFF  ((size_t)38 << 20)                    // bf16 [8192][1024]  16MB
#define VT_OFF  ((size_t)54 << 20)                    // bf16 [4][1024][2048] 16MB

// ---------------------------------------------------------------------------
// fp32 -> bf16 converts
// ---------------------------------------------------------------------------
__global__ void cvt_x_kernel(const float* __restrict__ X, __bf16* __restrict__ Xb, int n) {
    int i = blockIdx.x * blockDim.x + threadIdx.x;
    if (i < n) Xb[i] = f2bf(X[i]);
}

// W [C][H] row-major  ->  Wt [H][C] row-major (i.e. W transposed), 3 matrices
__global__ void cvt_w_kernel(const float* __restrict__ Wk, const float* __restrict__ Wq,
                             const float* __restrict__ Wv, __bf16* __restrict__ Wt3) {
    const float* W = (blockIdx.y == 0) ? Wk : (blockIdx.y == 1) ? Wq : Wv;
    __bf16* Wt = Wt3 + (size_t)blockIdx.y * (CC * HH);
    int i = blockIdx.x * blockDim.x + threadIdx.x;      // over C*H
    int k = i >> 10;            // row in W  (C / K dim)
    int n = i & 1023;           // col in W  (H / N dim)
    Wt[(size_t)n * CC + k] = f2bf(W[i]);
}

// ---------------------------------------------------------------------------
// Projection + LayerNorm:  Y = LN(Xb @ W + b) * g + beta
// grid.x = 512 (16-row blocks), grid.y = 3 (K / Q / V), 256 threads (8 waves)
// dynamic LDS: 16*1024 floats = 64KB
// ---------------------------------------------------------------------------
__global__ void proj_ln_kernel(const __bf16* __restrict__ Xb, const __bf16* __restrict__ Wt3,
                               const float* __restrict__ bk, const float* __restrict__ gk, const float* __restrict__ betak,
                               const float* __restrict__ bq, const float* __restrict__ gq, const float* __restrict__ betaq,
                               const float* __restrict__ bv, const float* __restrict__ gv, const float* __restrict__ betav,
                               __bf16* __restrict__ Kb, __bf16* __restrict__ Qb, __bf16* __restrict__ Vt) {
    extern __shared__ char smem[];
    float* Ysh = (float*)smem;                          // [16][1024]

    const int w       = threadIdx.x >> 5;
    const int lane    = threadIdx.x & 31;
    const int halfsel = lane >> 4;                      // 0: lanes 0-15, 1: lanes 16-31
    const int lane16  = lane & 15;
    const int proj    = blockIdx.y;
    const int m0      = blockIdx.x * 16;

    const __bf16* Wt   = Wt3 + (size_t)proj * (CC * HH);
    const float*  bias = (proj == 0) ? bk    : (proj == 1) ? bq    : bv;
    const float*  gain = (proj == 0) ? gk    : (proj == 1) ? gq    : gv;
    const float*  beta = (proj == 0) ? betak : (proj == 1) ? betaq : betav;

    const __bf16* Abase = Xb + (size_t)(m0 + lane16) * CC;
    const __bf16* Bcol  = Wt + (size_t)(w * 128 + lane16) * CC;   // per-lane column base
    const int colbase = w * 128;

    v8f acc[8];
    #pragma unroll
    for (int t = 0; t < 8; ++t) { v8f z = {0,0,0,0,0,0,0,0}; acc[t] = z; }

    #pragma unroll 1
    for (int kt = 0; kt < CC / 32; ++kt) {
        const int k0 = kt * 32;
        const int kA = k0 + halfsel * 8;                // A: K {0-7,16-23} or {8-15,24-31}
        const int kB = k0 + halfsel * 8;                // B: K {0-7,16-23} or {8-15,24-31}

        // issue ALL loads first (distinct regs) -> one wait, then 8 back-to-back WMMAs
        v16bf a = load_frag(Abase + kA, 16);
        v16bf bfr[8];
        #pragma unroll
        for (int t = 0; t < 8; ++t)
            bfr[t] = load_frag(Bcol + (size_t)t * 16 * CC + kB, 16);

        #pragma unroll
        for (int t = 0; t < 8; ++t)
            acc[t] = __builtin_amdgcn_wmma_f32_16x16x32_bf16(
                         false, a, false, bfr[t], (short)0, acc[t], false, false);
    }

    // C/D layout: element i -> row (i + 8*halfsel), col (tile*16 + lane16)
    #pragma unroll
    for (int t = 0; t < 8; ++t) {
        const int col = colbase + t * 16 + lane16;
        const float bc = bias[col];
        #pragma unroll
        for (int i = 0; i < 8; ++i) {
            const int r = i + 8 * halfsel;
            Ysh[r * HH + col] = acc[t][i] + bc;
        }
    }
    __syncthreads();

    // LayerNorm over H=1024: wave w owns rows 2w, 2w+1
    #pragma unroll 1
    for (int rr = 0; rr < 2; ++rr) {
        const int r = w * 2 + rr;
        float s = 0.f, s2 = 0.f;
        #pragma unroll 1
        for (int j = 0; j < HH / 32; ++j) {
            const float y = Ysh[r * HH + lane + 32 * j];
            s += y; s2 += y * y;
        }
        #pragma unroll
        for (int off = 16; off; off >>= 1) {
            s  += __shfl_xor(s,  off, 32);
            s2 += __shfl_xor(s2, off, 32);
        }
        const float mu  = s * (1.f / HH);
        const float var = s2 * (1.f / HH) - mu * mu;
        const float inv = rsqrtf(var + 1e-5f);
        const int m = m0 + r;
        #pragma unroll 1
        for (int j = 0; j < HH / 32; ++j) {
            const int col = lane + 32 * j;
            const float y = (Ysh[r * HH + col] - mu) * inv * gain[col] + beta[col];
            const __bf16 o = f2bf(y);
            if (proj == 0)      Kb[(size_t)m * HH + col] = o;
            else if (proj == 1) Qb[(size_t)m * HH + col] = o;
            else {              // V stored transposed per batch: Vt[b][c][l]
                const int b = m >> 11, l = m & (LL - 1);
                Vt[((size_t)b * CC + col) * LL + l] = o;
            }
        }
    }
}

// ---------------------------------------------------------------------------
// Attention: one workgroup per (batch, 16-row K block).
//   S[16][2048] = K_blk @ Q^T   (fp32, LDS)
//   softmax over 2048 cols      (fp32)
//   O = P @ V  (+ residual)     -> fp32 out
// dynamic LDS: 16*2048*4 + 16*2048*2 = 192KB
// ---------------------------------------------------------------------------
__global__ void attn_kernel(const __bf16* __restrict__ Kb, const __bf16* __restrict__ Qb,
                            const __bf16* __restrict__ Vt, const float* __restrict__ X,
                            float* __restrict__ Out) {
    extern __shared__ char smem[];
    float*  Ssh = (float*)smem;                                 // [16][2048] fp32
    __bf16* Psh = (__bf16*)(smem + (size_t)16 * LL * 4);        // [16][2048] bf16

    const int w       = threadIdx.x >> 5;
    const int lane    = threadIdx.x & 31;
    const int halfsel = lane >> 4;
    const int lane16  = lane & 15;
    const int b       = blockIdx.x >> 7;                        // 128 row-blocks / batch
    const int r0      = (blockIdx.x & 127) * 16;
    const int m0      = b * LL + r0;

    const __bf16* Abase = Kb + (size_t)(m0 + lane16) * CC;

    // ---- Phase 1: scores (each wave: 2 chunks of 128 cols = 8 tiles) ----
    #pragma unroll 1
    for (int chunk = 0; chunk < 2; ++chunk) {
        const int colbase = w * 256 + chunk * 128;
        const __bf16* Bcol = Qb + (size_t)(b * LL + colbase + lane16) * CC;
        v8f acc[8];
        #pragma unroll
        for (int t = 0; t < 8; ++t) { v8f z = {0,0,0,0,0,0,0,0}; acc[t] = z; }

        #pragma unroll 1
        for (int kt = 0; kt < CC / 32; ++kt) {
            const int k0 = kt * 32;
            const int kA = k0 + halfsel * 8;
            const int kB = k0 + halfsel * 8;

            v16bf a = load_frag(Abase + kA, 16);
            v16bf bfr[8];
            #pragma unroll
            for (int t = 0; t < 8; ++t)
                bfr[t] = load_frag(Bcol + (size_t)t * 16 * CC + kB, 16);

            #pragma unroll
            for (int t = 0; t < 8; ++t)
                acc[t] = __builtin_amdgcn_wmma_f32_16x16x32_bf16(
                             false, a, false, bfr[t], (short)0, acc[t], false, false);
        }
        #pragma unroll
        for (int t = 0; t < 8; ++t)
            #pragma unroll
            for (int i = 0; i < 8; ++i)
                Ssh[(i + 8 * halfsel) * LL + colbase + t * 16 + lane16] = acc[t][i];
    }
    __syncthreads();

    // ---- Phase 2: softmax over 2048 cols; wave w owns rows 2w, 2w+1 ----
    #pragma unroll 1
    for (int rr = 0; rr < 2; ++rr) {
        const int r = w * 2 + rr;
        float mx = -3.4e38f;
        #pragma unroll 1
        for (int j = 0; j < LL / 32; ++j)
            mx = fmaxf(mx, Ssh[r * LL + lane + 32 * j]);
        #pragma unroll
        for (int off = 16; off; off >>= 1)
            mx = fmaxf(mx, __shfl_xor(mx, off, 32));

        float s = 0.f;
        #pragma unroll 1
        for (int j = 0; j < LL / 32; ++j) {
            const int idx = r * LL + lane + 32 * j;
            const float e = __expf(Ssh[idx] - mx);
            Ssh[idx] = e;
            s += e;
        }
        #pragma unroll
        for (int off = 16; off; off >>= 1)
            s += __shfl_xor(s, off, 32);
        const float inv = 1.f / s;
        #pragma unroll 1
        for (int j = 0; j < LL / 32; ++j) {
            const int col = lane + 32 * j;
            Psh[r * LL + col] = f2bf(Ssh[r * LL + col] * inv);
        }
    }
    __syncthreads();

    // ---- Phase 3: O = P @ V; each wave owns 128 output channels ----
    const int cbase = w * 128;
    const __bf16* Bcol = Vt + ((size_t)b * CC + cbase + lane16) * LL;
    const __bf16* Prow = Psh + lane16 * LL;
    v8f acc[8];
    #pragma unroll
    for (int t = 0; t < 8; ++t) { v8f z = {0,0,0,0,0,0,0,0}; acc[t] = z; }

    #pragma unroll 1
    for (int kt = 0; kt < LL / 32; ++kt) {
        const int k0 = kt * 32;
        const int kA = k0 + halfsel * 8;
        const int kB = k0 + halfsel * 8;

        v16bf a = load_frag_lds(Prow + kA, 16);
        v16bf bfr[8];
        #pragma unroll
        for (int t = 0; t < 8; ++t)
            bfr[t] = load_frag(Bcol + (size_t)t * 16 * LL + kB, 16);

        #pragma unroll
        for (int t = 0; t < 8; ++t)
            acc[t] = __builtin_amdgcn_wmma_f32_16x16x32_bf16(
                         false, a, false, bfr[t], (short)0, acc[t], false, false);
    }

    // residual add + store fp32
    #pragma unroll
    for (int t = 0; t < 8; ++t) {
        #pragma unroll
        for (int i = 0; i < 8; ++i) {
            const int r = i + 8 * halfsel;
            const size_t idx = (size_t)(m0 + r) * CC + cbase + t * 16 + lane16;
            Out[idx] = acc[t][i] + X[idx];
        }
    }
}

// ---------------------------------------------------------------------------
extern "C" void kernel_launch(void* const* d_in, const int* in_sizes, int n_in,
                              void* d_out, int out_size, void* d_ws, size_t ws_size,
                              hipStream_t stream) {
    const float* X     = (const float*)d_in[0];
    const float* Wk    = (const float*)d_in[1];
    const float* bk    = (const float*)d_in[2];
    const float* gk    = (const float*)d_in[3];
    const float* betak = (const float*)d_in[4];
    const float* Wq    = (const float*)d_in[5];
    const float* bq    = (const float*)d_in[6];
    const float* gq    = (const float*)d_in[7];
    const float* betaq = (const float*)d_in[8];
    const float* Wv    = (const float*)d_in[9];
    const float* bv    = (const float*)d_in[10];
    const float* gv    = (const float*)d_in[11];
    const float* betav = (const float*)d_in[12];
    float* Out = (float*)d_out;

    char* ws = (char*)d_ws;
    __bf16* Xb  = (__bf16*)(ws + XB_OFF);
    __bf16* Wt3 = (__bf16*)(ws + WT_OFF);
    __bf16* Kb  = (__bf16*)(ws + KB_OFF);
    __bf16* Qb  = (__bf16*)(ws + QB_OFF);
    __bf16* Vt  = (__bf16*)(ws + VT_OFF);

    const int proj_smem = 16 * HH * 4;                  // 64 KB
    const int attn_smem = 16 * LL * 4 + 16 * LL * 2;    // 192 KB
    hipFuncSetAttribute((const void*)proj_ln_kernel,
                        hipFuncAttributeMaxDynamicSharedMemorySize, proj_smem);
    hipFuncSetAttribute((const void*)attn_kernel,
                        hipFuncAttributeMaxDynamicSharedMemorySize, attn_smem);

    // 1) convert input to bf16
    {
        const int n = MM * CC;
        cvt_x_kernel<<<n / 256, 256, 0, stream>>>(X, Xb, n);
    }
    // 2) convert + transpose weights to bf16
    {
        dim3 g(CC * HH / 256, 3);
        cvt_w_kernel<<<g, 256, 0, stream>>>(Wk, Wq, Wv, Wt3);
    }
    // 3) fused GEMM + bias + LayerNorm for K, Q, V
    {
        dim3 g(MM / 16, 3);
        proj_ln_kernel<<<g, 256, proj_smem, stream>>>(Xb, Wt3,
                                                      bk, gk, betak,
                                                      bq, gq, betaq,
                                                      bv, gv, betav,
                                                      Kb, Qb, Vt);
    }
    // 4) attention + residual
    {
        attn_kernel<<<BB * (LL / 16), 256, attn_smem, stream>>>(Kb, Qb, Vt, X, Out);
    }
}
